// NaiveDenseNetwork_27702539059326
// MI455X (gfx1250) — matrix operand order repaired and checked by reading, not compiled, not run on hardware
//
#include <hip/hip_runtime.h>
#include <hip/hip_bf16.h>
#include <stdint.h>

// ---------------------------------------------------------------------------
// Problem dims (fixed by the reference)
// ---------------------------------------------------------------------------
constexpr int B  = 256;
constexpr int L  = 64;
constexpr int E  = 128;
constexpr int S  = 16;
constexpr int H0 = 16 * E;   // 2048
constexpr int H1 = 4 * E;    // 512
constexpr int BL = B * L;    // 16384
constexpr int K0P = L * E + 32;   // 8224 = 8192 + 16 (sv) + 16 (zero pad), %32==0

typedef __attribute__((ext_vector_type(16))) _Float16 v16h;
typedef __attribute__((ext_vector_type(8)))  _Float16 v8h;
typedef __attribute__((ext_vector_type(4)))  _Float16 v4h;
typedef __attribute__((ext_vector_type(8)))  float    v8f;
typedef __attribute__((ext_vector_type(4)))  float    v4f;
typedef __attribute__((ext_vector_type(4)))  int      v4i;

#define AS1 __attribute__((address_space(1)))
#define AS3 __attribute__((address_space(3)))

// CDNA5 async load-to-LDS path (ASYNCcnt-tracked DMA into LDS)
#if defined(__has_builtin)
#  if __has_builtin(__builtin_amdgcn_global_load_async_to_lds_b128)
#    define USE_ASYNC_LDS 1
#  endif
#endif
#ifndef USE_ASYNC_LDS
#  define USE_ASYNC_LDS 0
#endif

__device__ __forceinline__ void wait_async_then_barrier() {
#if USE_ASYNC_LDS
#  if __has_builtin(__builtin_amdgcn_s_wait_asynccnt)
  __builtin_amdgcn_s_wait_asynccnt(0);
#  else
  asm volatile("s_wait_asynccnt 0x0" ::: "memory");
#  endif
#endif
  __syncthreads();
}

// ---------------------------------------------------------------------------
// Kernel 1: gather + norm-clip embedding rows -> f16 into A0 (B x K0P)
// ---------------------------------------------------------------------------
__global__ __launch_bounds__(128)
void k_emb(const int* __restrict__ tokens, const float* __restrict__ table,
           _Float16* __restrict__ A0) {
  const int bl = blockIdx.x;            // 0..16383
  const int b  = bl >> 6;
  const int l  = bl & 63;
  const int t  = threadIdx.x;           // 0..127
  const int tok = tokens[bl];
  const float e = table[(long)tok * E + t];
  float ss = e * e;
  #pragma unroll
  for (int off = 16; off > 0; off >>= 1) ss += __shfl_xor(ss, off, 32);
  __shared__ float red[4];
  if ((t & 31) == 0) red[t >> 5] = ss;
  __syncthreads();
  const float norm = sqrtf(red[0] + red[1] + red[2] + red[3]);
  const float scale = (norm > 1.0f) ? (1.0f / (norm + 1e-7f)) : 1.0f;
  A0[(long)b * K0P + l * E + t] = (_Float16)(e * scale);
}

// ---------------------------------------------------------------------------
// Kernel 2: sentiment vector + zero pad into A0 cols [8192, 8224)
// ---------------------------------------------------------------------------
__global__ __launch_bounds__(32)
void k_sv(const int* __restrict__ sids, const float* __restrict__ stab,
          _Float16* __restrict__ A0) {
  const int b = blockIdx.x;
  const int s = threadIdx.x;            // 0..31
  const float v = (s < S) ? stab[sids[b] * S + s] : 0.0f;
  A0[(long)b * K0P + L * E + s] = (_Float16)v;
}

__global__ __launch_bounds__(256)
void k_zero(float* __restrict__ p, int n) {
  const int i = blockIdx.x * 256 + threadIdx.x;
  if (i < n) p[i] = 0.0f;
}

__global__ __launch_bounds__(256)
void k_pred(const float* __restrict__ acc, const float* __restrict__ bp,
            float* __restrict__ out, int n) {
  const int i = blockIdx.x * 256 + threadIdx.x;
  if (i < n) out[i] = 1.0f / (1.0f + __expf(-(acc[i] + bp[0])));
}

// ---------------------------------------------------------------------------
// WMMA GEMM: C(MxN) = A(MxK, f16) * B(KxN, f32 global -> f16 LDS) [+epilogue]
//   MODE 0: base  = A0 @ W0[:8224]     -> +b0, store f32
//   MODE 1: h0    = emb @ W_word       -> +base[b], relu, store f16
//   MODE 2: h1    = h0 @ W1            -> +b1, relu, mask, dot Wp, atomicAdd
// Block tile 128x128x32, 8 waves (4x2), each wave 2x4 16x16x32 WMMA tiles.
// Double-buffered LDS, one barrier per K step. A tile staged by async DMA.
// ---------------------------------------------------------------------------
constexpr int BM = 128, BN = 128, BK = 32;
constexpr int LK = BK + 8;              // padded LDS row (f16) -> conflict-free b128

template<int MODE>
__device__ __forceinline__ long a_index(int m) {
  if constexpr (MODE == 0) return (long)m * K0P;                       // A0 rows
  if constexpr (MODE == 1) return (long)(m >> 6) * K0P + (m & 63) * E; // emb view of A0
  return (long)m * H0;                                                 // h0 rows
}

template<int MODE>
__global__ __launch_bounds__(256)
void gemm_k(const _Float16* __restrict__ A, const float* __restrict__ Bg, int ldb,
            int K,
            const float* __restrict__ bias,     // b0 (MODE 0) / b1 (MODE 2)
            const float* __restrict__ baseC,    // MODE 1
            _Float16* __restrict__ Oh,          // MODE 1 out (f16)
            float* __restrict__ Of,             // MODE 0 out (f32)
            const float* __restrict__ Wp,       // MODE 2
            const int* __restrict__ lens,       // MODE 2
            float* __restrict__ acc_pred)       // MODE 2
{
  __shared__ __align__(16) _Float16 As[2][BM * LK];
  __shared__ __align__(16) _Float16 Bs[2][BN * LK];

  const int tid  = threadIdx.x;
  const int lane = tid & 31;
  const int wave = tid >> 5;
  const int wm = wave >> 1;             // 0..3 (rows)
  const int wn = wave & 1;              // 0..1 (cols, 64 each)
  const int m0 = blockIdx.y * BM;
  const int n0 = blockIdx.x * BN;

  v8f acc[2][4] = {};

  // A loader: 128x32 f16, 2 x b128 per thread
  const int ar = tid >> 2;              // 0..63
  const int ac = (tid & 3) * 8;         // 0,8,16,24
  // B loader: 32k x 128n f32 -> Bs[n][k] f16; thread owns 4 k x 4 n
  const int bkq = (tid >> 5) * 4;       // 0,4,...,28
  const int bnq = (tid & 31) * 4;       // 0..124

  auto stage = [&](int k0, int buf) {
    _Float16* __restrict__ Asb = As[buf];
    _Float16* __restrict__ Bsb = Bs[buf];
    // ---- A tile: straight f16 copy -> async DMA to LDS when available ----
    #pragma unroll
    for (int h = 0; h < 2; ++h) {
      const int row = ar + h * 64;
      const _Float16* gsrc = A + a_index<MODE>(m0 + row) + k0 + ac;
      _Float16* ldst = &Asb[row * LK + ac];
#if USE_ASYNC_LDS
      __builtin_amdgcn_global_load_async_to_lds_b128(
          (AS1 v4i*)(unsigned long long)(uintptr_t)gsrc,
          (AS3 v4i*)(unsigned int)(uintptr_t)ldst,
          0, 0);
#else
      *(v8h*)ldst = *(const v8h*)gsrc;
#endif
    }
    // ---- B tile: f32 -> f16 convert + transpose (VGPR path) ----
    const float* src = Bg + (long)(k0 + bkq) * ldb + n0 + bnq;
    const v4f r0 = *(const v4f*)(src);
    const v4f r1 = *(const v4f*)(src + ldb);
    const v4f r2 = *(const v4f*)(src + 2 * (long)ldb);
    const v4f r3 = *(const v4f*)(src + 3 * (long)ldb);
    #pragma unroll
    for (int j = 0; j < 4; ++j) {
      v4h p;
      p[0] = (_Float16)r0[j]; p[1] = (_Float16)r1[j];
      p[2] = (_Float16)r2[j]; p[3] = (_Float16)r3[j];
      *(v4h*)(&Bsb[(bnq + j) * LK + bkq]) = p;   // ds_store_b64
    }
  };

  stage(0, 0);
  wait_async_then_barrier();

  int buf = 0;
  for (int k0 = 0; k0 < K; k0 += BK) {
    const int nb = buf ^ 1;
    if (k0 + BK < K) stage(k0 + BK, nb);        // overlap with compute

    const _Float16* __restrict__ Asb = As[buf];
    const _Float16* __restrict__ Bsb = Bs[buf];
    const int khalf = (lane >> 4) * 8;   // A frag: runs at khalf, khalf+16
    const int kbase = (lane >> 4) * 8;   // B frag: runs at kbase, kbase+16
    #pragma unroll
    for (int mt = 0; mt < 2; ++mt) {
      const _Float16* ap = &Asb[(wm * 32 + mt * 16 + (lane & 15)) * LK];
      const v8h alo = *(const v8h*)(ap + khalf);
      const v8h ahi = *(const v8h*)(ap + khalf + 16);
      const v16h af = __builtin_shufflevector(alo, ahi,
          0,1,2,3,4,5,6,7,8,9,10,11,12,13,14,15);
      #pragma unroll
      for (int nt = 0; nt < 4; ++nt) {
        const _Float16* bp2 = &Bsb[(wn * 64 + nt * 16 + (lane & 15)) * LK + kbase];
        const v8h blo = *(const v8h*)(bp2);
        const v8h bhi = *(const v8h*)(bp2 + 16);
        const v16h bf = __builtin_shufflevector(blo, bhi,
            0,1,2,3,4,5,6,7,8,9,10,11,12,13,14,15);
        acc[mt][nt] = __builtin_amdgcn_wmma_f32_16x16x32_f16(
            false, af, false, bf, (short)0, acc[mt][nt], false, false);
      }
    }
    wait_async_then_barrier();
    buf = nb;
  }

  // ---- epilogues ----
  const int rlane = (lane >> 4) * 8;    // row offset within 16x16 tile
  const int clane = lane & 15;          // col within tile
  const int mtb = m0 + wm * 32;
  const int ntb = n0 + wn * 64;

  if constexpr (MODE == 0) {
    #pragma unroll
    for (int mt = 0; mt < 2; ++mt)
      #pragma unroll
      for (int nt = 0; nt < 4; ++nt) {
        const int n = ntb + nt * 16 + clane;
        const float bv = bias[n];
        #pragma unroll
        for (int i = 0; i < 8; ++i) {
          const int m = mtb + mt * 16 + rlane + i;
          Of[(long)m * H0 + n] = acc[mt][nt][i] + bv;
        }
      }
  } else if constexpr (MODE == 1) {
    #pragma unroll
    for (int mt = 0; mt < 2; ++mt)
      #pragma unroll
      for (int nt = 0; nt < 4; ++nt) {
        const int n = ntb + nt * 16 + clane;
        #pragma unroll
        for (int i = 0; i < 8; ++i) {
          const int m = mtb + mt * 16 + rlane + i;
          float v = acc[mt][nt][i] + baseC[(long)(m >> 6) * H0 + n];
          v = fmaxf(v, 0.0f);
          Oh[(long)m * H0 + n] = (_Float16)v;
        }
      }
  } else {
    float bv[4], wv[4];
    #pragma unroll
    for (int nt = 0; nt < 4; ++nt) {
      const int n = ntb + nt * 16 + clane;
      bv[nt] = bias[n];
      wv[nt] = Wp[n];
    }
    #pragma unroll
    for (int mt = 0; mt < 2; ++mt) {
      #pragma unroll
      for (int i = 0; i < 8; ++i) {
        float s = 0.0f;
        #pragma unroll
        for (int nt = 0; nt < 4; ++nt)
          s += fmaxf(acc[mt][nt][i] + bv[nt], 0.0f) * wv[nt];
        #pragma unroll
        for (int off = 1; off < 16; off <<= 1) s += __shfl_xor(s, off, 16);
        if (clane == 0) {
          const int m = mtb + mt * 16 + rlane + i;
          const int bb = m >> 6, ll = m & 63;
          if (ll < lens[bb]) atomicAdd(&acc_pred[m], s);
        }
      }
    }
  }
}

// ---------------------------------------------------------------------------
// Launch
// ---------------------------------------------------------------------------
extern "C" void kernel_launch(void* const* d_in, const int* in_sizes, int n_in,
                              void* d_out, int out_size, void* d_ws, size_t ws_size,
                              hipStream_t stream) {
  const int*   text_batch  = (const int*)  d_in[0];
  const int*   text_len    = (const int*)  d_in[1];
  const int*   sent_ids    = (const int*)  d_in[2];
  const float* emb_table   = (const float*)d_in[3];
  const float* sent_table  = (const float*)d_in[4];
  const float* W0          = (const float*)d_in[5];
  const float* b0          = (const float*)d_in[6];
  const float* W1          = (const float*)d_in[7];
  const float* b1          = (const float*)d_in[8];
  const float* Wp          = (const float*)d_in[9];
  const float* bp          = (const float*)d_in[10];
  float* out = (float*)d_out;

  // workspace carve-up (all offsets 256B aligned)
  char* ws = (char*)d_ws;
  _Float16* A0   = (_Float16*)ws;                 ws += (size_t)B * K0P * 2;   // 4,210,688
  float*    base = (float*)ws;                    ws += (size_t)B * H0 * 4;    // 2,097,152
  _Float16* h0   = (_Float16*)ws;                 ws += (size_t)BL * H0 * 2;   // 67,108,864
  float*    pacc = (float*)ws;                    ws += (size_t)BL * 4;        //    65,536

  k_emb<<<dim3(BL), dim3(128), 0, stream>>>(text_batch, emb_table, A0);
  k_sv <<<dim3(B),  dim3(32),  0, stream>>>(sent_ids, sent_table, A0);
  k_zero<<<dim3((BL + 255) / 256), dim3(256), 0, stream>>>(pacc, BL);

  // base = A0 @ W0[:8224] + b0         (M=256, N=2048, K=8224)
  gemm_k<0><<<dim3(H0 / BN, B / BM), dim3(256), 0, stream>>>(
      A0, W0, H0, K0P, b0, nullptr, nullptr, base, nullptr, nullptr, nullptr);

  // h0 = relu(base + emb @ W_word)     (M=16384, N=2048, K=128)
  gemm_k<1><<<dim3(H0 / BN, BL / BM), dim3(256), 0, stream>>>(
      A0, W0 + (size_t)(L * E + S) * H0, H0, E, nullptr, base, h0,
      nullptr, nullptr, nullptr, nullptr);

  // h1 = relu(h0 @ W1 + b1); masked dot with Wp -> pacc
  gemm_k<2><<<dim3(H1 / BN, BL / BM), dim3(256), 0, stream>>>(
      h0, W1, H1, H0, b1, nullptr, nullptr, nullptr, Wp, text_len, pacc);

  // pred = sigmoid(pacc + bp)
  k_pred<<<dim3((BL + 255) / 256), dim3(256), 0, stream>>>(pacc, bp, out, BL);
}
